// GatedLinearAttention_1108101562755
// MI455X (gfx1250) — matrix operand order repaired and checked by reading, not compiled, not run on hardware
//
#include <hip/hip_runtime.h>
#include <hip/hip_bf16.h>

// ---------------------------------------------------------------------------
// Gated linear attention for MI455X (gfx1250, wave32, WMMA).
// B=2, S=8192, H=1024, NH=16, dk=dv=64.
// Pipeline:
//   1) f32 -> bf16: X; transpose-convert Wq|Wk|Wv|Wg -> Wcat^T [4096x1024],
//      Wo -> Wo^T [1024x1024]
//   2) TN GEMM bf16 (v_wmma_f32_16x16x32_bf16): qkvg = X @ Wcat  [16384x4096]
//      hot loop: async global->LDS (ASYNCcnt) double buffer feeding WMMA
//   3) k_sum over sequence (elu+1 applied on the fly)
//   4) per-token: qk, normalizer, z, LayerNorm, SiLU gate -> Z bf16
//   5) TN GEMM bf16: out = Z @ Wo   [16384 x 1024] fp32
// ---------------------------------------------------------------------------

typedef __attribute__((ext_vector_type(16))) __bf16    v16bf;
typedef __attribute__((ext_vector_type(8)))  float     v8f;
typedef __attribute__((ext_vector_type(4)))  unsigned  u32x4;
typedef __attribute__((ext_vector_type(2)))  unsigned  u32x2;
typedef __attribute__((ext_vector_type(4)))  int       v4i;

union FragU { u32x4 u[2]; v16bf bf; };

#define NUM_HEADS 16
#define BSZ 2
#define SEQ 8192
#define HID 1024
#define TOK (BSZ * SEQ)       // 16384
#define HCAT 4096             // q|k|v|g concatenated

#if defined(__gfx1250__) && __has_builtin(__builtin_amdgcn_global_load_async_to_lds_b128)
#define USE_ASYNC_LDS 1
#else
#define USE_ASYNC_LDS 0
#endif

__device__ __forceinline__ unsigned short f2bf(float f) {
    unsigned u = __float_as_uint(f);
    u += 0x7FFFu + ((u >> 16) & 1u);   // round-to-nearest-even
    return (unsigned short)(u >> 16);
}

__device__ __forceinline__ float elu1(float x) {
    return x > 0.f ? x + 1.f : __expf(x);
}

// --------------------------- conversion kernels ----------------------------

__global__ __launch_bounds__(256) void cvt_bf16_kernel(
    const float* __restrict__ src, unsigned short* __restrict__ dst, size_t n) {
    size_t i = (size_t)blockIdx.x * 256u + threadIdx.x;
    if (i < n) dst[i] = f2bf(src[i]);
}

// src: [K][cols] row-major f32  ->  dst: [cols][K] row-major bf16 (transposed)
__global__ __launch_bounds__(256) void cvt_transpose_kernel(
    const float* __restrict__ src, unsigned short* __restrict__ dst,
    int cols, int K) {
    int idx = blockIdx.x * 256 + threadIdx.x;   // over K*cols
    int k = idx / cols, n = idx - k * cols;
    dst[(size_t)n * K + k] = f2bf(src[idx]);
}

// ------------------------------ bf16 WMMA GEMM (TN) ------------------------
// C[M,N] (f32) = A[M,K] (bf16 row-major) x Bt[N,K]^T (bf16 N-major).
// Block tile 128x128x64, 256 threads = 8 waves, wave tile 32x64 (2x4 WMMA).

#define BM 128
#define BN 128
#define BK 64
#define LDSS 72   // padded LDS row stride (elems) = 144 B; 16B-aligned, conflict-free

// one K-tile of WMMA work for one wave (2 A-frags x 4 B-frags x 2 k-subtiles)
__device__ __forceinline__ void wmma_tile(
    const unsigned short* __restrict__ As, const unsigned short* __restrict__ Bs,
    int wm, int wn, int lrow, int lhi, v8f acc[2][4]) {
#pragma unroll
    for (int kk = 0; kk < BK; kk += 32) {
        // A fragments 16x32: lanes 0-15 hold K 0-7 (v0-3) & 16-23 (v4-7),
        // lanes 16-31 hold K 8-15 & 24-31.
        v16bf afrag[2];
#pragma unroll
        for (int i = 0; i < 2; ++i) {
            const unsigned short* p = As + (wm * 32 + i * 16 + lrow) * LDSS + kk;
            FragU f;
            f.u[0] = *(const u32x4*)(p + lhi * 8);
            f.u[1] = *(const u32x4*)(p + 16 + lhi * 8);
            afrag[i] = f.bf;
        }
        // B fragments 32x16: lane = column n; lanes 0-15 hold K 0-7 & 16-23,
        // lanes 16-31 hold K 8-15 & 24-31 (same order as A).
        v16bf bfrag[4];
#pragma unroll
        for (int j = 0; j < 4; ++j) {
            const unsigned short* p =
                Bs + (wn * 64 + j * 16 + lrow) * LDSS + kk + lhi * 8;
            FragU f;
            f.u[0] = *(const u32x4*)(p);
            f.u[1] = *(const u32x4*)(p + 16);
            bfrag[j] = f.bf;
        }
#pragma unroll
        for (int i = 0; i < 2; ++i)
#pragma unroll
            for (int j = 0; j < 4; ++j)
                acc[i][j] = __builtin_amdgcn_wmma_f32_16x16x32_bf16(
                    false, afrag[i], false, bfrag[j],
                    (short)0, acc[i][j], false, false);
    }
}

__global__ __launch_bounds__(256) void gemm_bf16_tn_kernel(
    const unsigned short* __restrict__ A,    // [M][K]
    const unsigned short* __restrict__ Bt,   // [N][K]
    float* __restrict__ C,                   // [M][N]
    int M, int N, int K) {

    const int tid  = threadIdx.x;
    const int lane = tid & 31;
    const int wave = tid >> 5;
    const int wm   = wave & 3;     // 4 waves along M
    const int wn   = wave >> 2;    // 2 waves along N
    const int lrow = lane & 15;
    const int lhi  = lane >> 4;

    const int m0 = blockIdx.y * BM;
    const int n0 = blockIdx.x * BN;

    v8f acc[2][4];
#pragma unroll
    for (int i = 0; i < 2; ++i)
#pragma unroll
        for (int j = 0; j < 4; ++j)
#pragma unroll
            for (int e = 0; e < 8; ++e) acc[i][j][e] = 0.f;

    // staging: each thread owns one 64-byte half-row of each tile
    const int r = tid >> 1;            // 0..127 tile row
    const int h = (tid & 1) * 32;      // elem offset within row (0 or 32)

    const unsigned short* Aptr = A  + (size_t)(m0 + r) * K + h;
    const unsigned short* Bptr = Bt + (size_t)(n0 + r) * K + h;

#if USE_ASYNC_LDS
    // -------- async global->LDS double buffer (ASYNCcnt, no VGPR staging) ---
    __shared__ unsigned short As[2][BM * LDSS];
    __shared__ unsigned short Bs[2][BN * LDSS];

    unsigned short* la = &As[0][r * LDSS + h];
    unsigned short* lb = &Bs[0][r * LDSS + h];
    const int bufstride = BM * LDSS;   // elems between buffer 0 and 1

    // issue one K-tile (8 async b128 per thread: 4 A + 4 B)
    auto issue = [&](int t, int buf) {
        const unsigned short* ga = Aptr + t * BK;
        const unsigned short* gb = Bptr + t * BK;
        unsigned short* da = la + buf * bufstride;
        unsigned short* db = lb + buf * bufstride;
#pragma unroll
        for (int j = 0; j < 4; ++j) {
            __builtin_amdgcn_global_load_async_to_lds_b128(
                (v4i*)(ga + 8 * j), (v4i*)(da + 8 * j), 0, 0);
            __builtin_amdgcn_global_load_async_to_lds_b128(
                (v4i*)(gb + 8 * j), (v4i*)(db + 8 * j), 0, 0);
        }
    };

    const int T = K / BK;
    issue(0, 0);
    if (T > 1) issue(1, 1);

    for (int t = 0; t < T; ++t) {
        // tile t is this wave's oldest 8 async ops (in-order counter)
        if (t + 1 < T) asm volatile("s_wait_asynccnt 0x8" ::: "memory");
        else           asm volatile("s_wait_asynccnt 0x0" ::: "memory");
        __syncthreads();                       // buf[t&1] ready for all waves

        wmma_tile(As[t & 1], Bs[t & 1], wm, wn, lrow, lhi, acc);

        __syncthreads();                       // all waves done reading buf[t&1]
        if (t + 2 < T) issue(t + 2, t & 1);    // refill the buffer just freed
    }
#else
    // -------- fallback: register-prefetch pipeline --------------------------
    __shared__ unsigned short As[BM * LDSS];
    __shared__ unsigned short Bs[BN * LDSS];

    u32x4 ra[4], rb[4];
#pragma unroll
    for (int j = 0; j < 4; ++j) {
        ra[j] = *(const u32x4*)(Aptr + 8 * j);
        rb[j] = *(const u32x4*)(Bptr + 8 * j);
    }
    unsigned short* asw = As + r * LDSS + h;
    unsigned short* bsw = Bs + r * LDSS + h;

    for (int k0 = 0; k0 < K; k0 += BK) {
        __syncthreads();
#pragma unroll
        for (int j = 0; j < 4; ++j) {
            *(u32x4*)(asw + 8 * j) = ra[j];
            *(u32x4*)(bsw + 8 * j) = rb[j];
        }
        __syncthreads();
        if (k0 + BK < K) {
            Aptr += BK; Bptr += BK;
#pragma unroll
            for (int j = 0; j < 4; ++j) {
                ra[j] = *(const u32x4*)(Aptr + 8 * j);
                rb[j] = *(const u32x4*)(Bptr + 8 * j);
            }
        }
        wmma_tile(As, Bs, wm, wn, lrow, lhi, acc);
    }
#endif

    // D layout: lane holds column n = lane&15; lanes 0-15 rows 0-7 in v0-7,
    // lanes 16-31 rows 8-15.
#pragma unroll
    for (int i = 0; i < 2; ++i) {
        const int mrow = m0 + wm * 32 + i * 16 + lhi * 8;
#pragma unroll
        for (int j = 0; j < 4; ++j) {
            const int c = n0 + wn * 64 + j * 16 + lrow;
            float* cp = C + (size_t)mrow * N + c;
#pragma unroll
            for (int rr = 0; rr < 8; ++rr) cp[(size_t)rr * N] = acc[i][j][rr];
        }
    }
}

// ------------------------------ k_sum kernels ------------------------------

__global__ __launch_bounds__(256) void ksum_zero_kernel(float* ksum) {
    ksum[blockIdx.x * 256 + threadIdx.x] = 0.f;
}

// grid: (HID/256, S-chunks, B); each thread owns one column, 256-row slab.
__global__ __launch_bounds__(256) void ksum_kernel(
    const float* __restrict__ qkvg, float* __restrict__ ksum) {
    const int c  = blockIdx.x * 256 + threadIdx.x;  // 0..1023
    const int b  = blockIdx.z;
    const int s0 = blockIdx.y * 256;
    const float* base = qkvg + ((size_t)b * SEQ + s0) * HCAT + HID + c;
    float acc = 0.f;
    for (int s = 0; s < 256; ++s) acc += elu1(base[(size_t)s * HCAT]);
    atomicAdd(&ksum[b * HID + c], acc);
}

// --------------------------- per-token epilogue ----------------------------
// One block per token, 256 threads; thread t owns columns 4t..4t+3 (all in
// one head: head = t/16; 16-lane shfl groups align with head boundaries).

__global__ __launch_bounds__(256) void token_kernel(
    const float* __restrict__ qkvg, const float* __restrict__ ksum,
    const float* __restrict__ gamma, const float* __restrict__ beta,
    unsigned short* __restrict__ zb) {

    const int t   = blockIdx.x;
    const int b   = t / SEQ;
    const int tid = threadIdx.x;
    const int c   = tid * 4;
    const float* row = qkvg + (size_t)t * HCAT;

    float4 q  = *(const float4*)(row + c);
    float4 k  = *(const float4*)(row + HID + c);
    float4 v  = *(const float4*)(row + 2 * HID + c);
    float4 g  = *(const float4*)(row + 3 * HID + c);
    float4 ks = *(const float4*)(ksum + b * HID + c);

    float4 qe = make_float4(elu1(q.x), elu1(q.y), elu1(q.z), elu1(q.w));
    float4 ke = make_float4(elu1(k.x), elu1(k.y), elu1(k.z), elu1(k.w));

    float qk = qe.x * ke.x + qe.y * ke.y + qe.z * ke.z + qe.w * ke.w;
    float nm = qe.x * ks.x + qe.y * ks.y + qe.z * ks.z + qe.w * ks.w;
#pragma unroll
    for (int m = 1; m < 16; m <<= 1) {
        qk += __shfl_xor(qk, m, 32);
        nm += __shfl_xor(nm, m, 32);
    }
    const float sc = qk / (nm + 1e-6f);
    float z[4] = {sc * v.x, sc * v.y, sc * v.z, sc * v.w};

    // LayerNorm block reduction over 1024 values
    float s1 = z[0] + z[1] + z[2] + z[3];
    float s2 = z[0] * z[0] + z[1] * z[1] + z[2] * z[2] + z[3] * z[3];
#pragma unroll
    for (int m = 1; m < 32; m <<= 1) {
        s1 += __shfl_xor(s1, m, 32);
        s2 += __shfl_xor(s2, m, 32);
    }
    __shared__ float r1[8], r2[8];
    const int lane = tid & 31, wave = tid >> 5;
    if (lane == 0) { r1[wave] = s1; r2[wave] = s2; }
    __syncthreads();
    float S1 = 0.f, S2 = 0.f;
#pragma unroll
    for (int w = 0; w < 8; ++w) { S1 += r1[w]; S2 += r2[w]; }
    const float mu   = S1 * (1.f / HID);
    const float var  = S2 * (1.f / HID) - mu * mu;
    const float rstd = rsqrtf(var + 1e-5f);

    float4 ga = *(const float4*)(gamma + c);
    float4 be = *(const float4*)(beta + c);
    float gv[4] = {g.x, g.y, g.z, g.w};
    float gm[4] = {ga.x, ga.y, ga.z, ga.w};
    float bt[4] = {be.x, be.y, be.z, be.w};

    u32x2 packed;
#pragma unroll
    for (int j = 0; j < 2; ++j) {
        unsigned lo, hi;
        {
            int e = 2 * j;
            float zn = (z[e] - mu) * rstd * gm[e] + bt[e];
            float sg = gv[e] / (1.f + __expf(-gv[e]));
            lo = f2bf(zn * sg);
        }
        {
            int e = 2 * j + 1;
            float zn = (z[e] - mu) * rstd * gm[e] + bt[e];
            float sg = gv[e] / (1.f + __expf(-gv[e]));
            hi = f2bf(zn * sg);
        }
        packed[j] = lo | (hi << 16);
    }
    *(u32x2*)(zb + (size_t)t * HID + c) = packed;
}

// ------------------------------- launcher ----------------------------------

extern "C" void kernel_launch(void* const* d_in, const int* in_sizes, int n_in,
                              void* d_out, int out_size, void* d_ws, size_t ws_size,
                              hipStream_t stream) {
    const float* X     = (const float*)d_in[0];
    const float* Wq    = (const float*)d_in[1];
    const float* Wk    = (const float*)d_in[2];
    const float* Wv    = (const float*)d_in[3];
    const float* Wg    = (const float*)d_in[4];
    const float* Wo    = (const float*)d_in[5];
    const float* gamma = (const float*)d_in[6];
    const float* beta  = (const float*)d_in[7];
    float* out = (float*)d_out;

    // workspace layout (bytes)
    char* ws = (char*)d_ws;
    unsigned short* Xb   = (unsigned short*)(ws);                 // 32 MB
    unsigned short* Wct  = (unsigned short*)(ws + 33554432ull);   //  8 MB  Wcat^T [4096][1024]
    unsigned short* Wot  = (unsigned short*)(ws + 41943040ull);   //  2 MB  Wo^T   [1024][1024]
    float*          qkvg = (float*)        (ws + 44040192ull);    // 256 MB
    float*          ksum = (float*)        (ws + 312475648ull);   //  8 KB
    unsigned short* zb   = (unsigned short*)(ws + 312483840ull);  // 32 MB

    // 1) conversions (weights transposed once: GEMM hot loop stays scatter-free)
    cvt_bf16_kernel<<<(TOK * HID) / 256, 256, 0, stream>>>(X, Xb, (size_t)TOK * HID);
    cvt_transpose_kernel<<<(HID * HID) / 256, 256, 0, stream>>>(Wq, Wct + 0ull * HID * HID, HID, HID);
    cvt_transpose_kernel<<<(HID * HID) / 256, 256, 0, stream>>>(Wk, Wct + 1ull * HID * HID, HID, HID);
    cvt_transpose_kernel<<<(HID * HID) / 256, 256, 0, stream>>>(Wv, Wct + 2ull * HID * HID, HID, HID);
    cvt_transpose_kernel<<<(HID * HID) / 256, 256, 0, stream>>>(Wg, Wct + 3ull * HID * HID, HID, HID);
    cvt_transpose_kernel<<<(HID * HID) / 256, 256, 0, stream>>>(Wo, Wot, HID, HID);

    // 2) fused q|k|v|g projection GEMM: [16384,1024] x [1024,4096]
    gemm_bf16_tn_kernel<<<dim3(HCAT / BN, TOK / BM), 256, 0, stream>>>(
        Xb, Wct, qkvg, TOK, HCAT, HID);

    // 3) k_sum over sequence
    ksum_zero_kernel<<<(BSZ * HID) / 256, 256, 0, stream>>>(ksum);
    ksum_kernel<<<dim3(HID / 256, SEQ / 256, BSZ), 256, 0, stream>>>(qkvg, ksum);

    // 4) per-token attention math + LayerNorm + SiLU gate -> bf16 Z
    token_kernel<<<TOK, 256, 0, stream>>>(qkvg, ksum, gamma, beta, zb);

    // 5) output projection GEMM: [16384,1024] x [1024,1024] -> fp32 out
    gemm_bf16_tn_kernel<<<dim3(HID / BN, TOK / BM), 256, 0, stream>>>(
        zb, Wot, out, TOK, HID, HID);
}
